// GAT_medium_6201932775763
// MI455X (gfx1250) — matrix operand order repaired and checked by reading, not compiled, not run on hardware
//
#include <hip/hip_runtime.h>

// ---------------------------------------------------------------------------
// Types for CDNA5 WMMA (wave32, v_wmma_f32_16x16x32_bf16)
// ---------------------------------------------------------------------------
typedef __attribute__((ext_vector_type(16))) __bf16          v16bf;
typedef __attribute__((ext_vector_type(16))) unsigned short  v16us;
typedef __attribute__((ext_vector_type(8)))  float           v8f;

#define NEG_SLOPE 0.2f

#if defined(__gfx1250__) && __has_builtin(__builtin_amdgcn_global_load_async_to_lds_b128)
#define HAVE_ASYNC_LDS 1
#endif

// fp32 -> bf16 (round-to-nearest-even, ignoring NaN corner case)
__device__ __forceinline__ unsigned short f2bf(float f) {
  unsigned int u = __float_as_uint(f);
  unsigned int r = u + 0x7FFFu + ((u >> 16) & 1u);
  return (unsigned short)(r >> 16);
}

__device__ __forceinline__ void atomicMaxF(float* addr, float val) {
  unsigned int* ua = (unsigned int*)addr;
  unsigned int old = *ua;
  while (__uint_as_float(old) < val) {
    unsigned int assumed = old;
    old = atomicCAS(ua, assumed, __float_as_uint(val));
    if (old == assumed) break;
  }
}

// ---------------------------------------------------------------------------
// Fragment-order helpers (ISA 7.12.2, 16-bit operands, wave32)
// A 16x32 tile: lane = hi*16 + r (r=row), elem j<8 -> K=hi*8+j ; j>=8 -> K=8+hi*8+j
//   inverse: km=K%32: hi=(km>>3)&1, j=(km&7)+((km&16)?8:0)
// B 32x16 tile: lane = hi*16 + n (n=col), elem j<8 -> K=hi*8+j ; j>=8 -> K=8+hi*8+j
//   inverse: hi=(km>>3)&1, j=(km&7)+((km&16)?8:0)
// ---------------------------------------------------------------------------

// Pack fp32 features [n x Kdim] (optionally +bias, relu) into A-fragment bf16
__global__ void pack_feat_kernel(const float* __restrict__ src, const float* __restrict__ bias,
                                 unsigned short* __restrict__ dst, long nelem, int Kdim,
                                 int relu) {
  long idx = (long)blockIdx.x * blockDim.x + threadIdx.x;
  if (idx >= nelem) return;
  int n = (int)(idx / Kdim), k = (int)(idx % Kdim);
  float v = src[idx];
  if (bias) v += bias[k];
  if (relu) v = v > 0.f ? v : 0.f;
  int nkb = Kdim >> 5;
  int t = n >> 4, r = n & 15, kb = k >> 5, km = k & 31;
  int hi = (km >> 3) & 1, j = (km & 7) + ((km & 16) ? 8 : 0);
  dst[(((long)t * nkb + kb) * 32 + hi * 16 + r) * 16 + j] = f2bf(v);
}

// Pack fp32 weights [K x OUTD] into B-fragment bf16
__global__ void pack_w_kernel(const float* __restrict__ W, unsigned short* __restrict__ dst,
                              int K, int OUTD) {
  int idx = blockIdx.x * blockDim.x + threadIdx.x;
  if (idx >= K * OUTD) return;
  int k = idx / OUTD, c = idx % OUTD;
  int nkb = K >> 5;
  int ct = c >> 4, n = c & 15, kb = k >> 5, km = k & 31;
  int hi = (km >> 3) & 1, j = (km & 7) + ((km & 16) ? 8 : 0);
  dst[((ct * nkb + kb) * 32 + hi * 16 + n) * 16 + j] = f2bf(W[idx]);
}

// ---------------------------------------------------------------------------
// bf16 WMMA GEMM on pre-swizzled operands.
//   Asw: [ntiles(rows)][K/32][32][16] bf16   Bsw: [OUTD/16][K/32][32][16] bf16
// Block = 128 threads (4 waves), one 16-row tile per block; each wave owns
// col-tiles {w, w+4}. A tile staged to LDS (async copy when available) and
// shared by all 4 waves; B read straight from global (L2-resident).
// ---------------------------------------------------------------------------
__global__ void gemm_swz_bf16_wmma(const unsigned short* __restrict__ Asw,
                                   const unsigned short* __restrict__ Bsw,
                                   const float* __restrict__ bias, float* __restrict__ Cout,
                                   int nrows, int K, int OUTD, int relu) {
  __shared__ __align__(32) unsigned short Atile[16 * 128];  // up to 4KB used
  const int tid = threadIdx.x;
  const int nkb = K >> 5;
  const unsigned short* Ablk = Asw + (size_t)blockIdx.x * (size_t)(16 * K);

  // ---- stage A tile (raw byte copy; layout already fragment-ordered) ----
  {
    const int nch = (16 * K) / 8;  // 16-byte chunks
#if defined(HAVE_ASYNC_LDS)
    typedef int v4i_vs __attribute__((vector_size(16)));
    typedef __attribute__((address_space(1))) v4i_vs* gp1;
    typedef __attribute__((address_space(3))) v4i_vs* lp3;
    for (int i = tid; i < nch; i += 128) {
      gp1 g = (gp1)(unsigned long long)(const void*)(Ablk + i * 8);
      lp3 l = (lp3)(unsigned int)(unsigned long long)(void*)(&Atile[i * 8]);
      __builtin_amdgcn_global_load_async_to_lds_b128(g, l, 0, 0);
    }
#if __has_builtin(__builtin_amdgcn_s_wait_asynccnt)
    __builtin_amdgcn_s_wait_asynccnt(0);
#else
    asm volatile("s_wait_asynccnt 0x0" ::: "memory");
#endif
#else
    for (int i = tid; i < nch; i += 128)
      ((uint4*)Atile)[i] = ((const uint4*)Ablk)[i];
#endif
  }
  __syncthreads();

  const int wave = tid >> 5, lane = tid & 31;
  const int r = lane & 15, hi = lane >> 4;
  const int ntile = OUTD >> 4;
  const bool two = (ntile > 4);
  const int ct0 = wave, ct1 = wave + 4;

  v8f acc0 = {}, acc1 = {};
  for (int kb = 0; kb < nkb; ++kb) {
    v16us au = *(const v16us*)&Atile[(kb * 32 + lane) * 16];
    v16bf a = __builtin_bit_cast(v16bf, au);
    v16us bu0 = *(const v16us*)&Bsw[((ct0 * nkb + kb) * 32 + lane) * 16];
    acc0 = __builtin_amdgcn_wmma_f32_16x16x32_bf16(
        false, a, false, __builtin_bit_cast(v16bf, bu0), (short)0, acc0, false, false);
    if (two) {
      v16us bu1 = *(const v16us*)&Bsw[((ct1 * nkb + kb) * 32 + lane) * 16];
      acc1 = __builtin_amdgcn_wmma_f32_16x16x32_bf16(
          false, a, false, __builtin_bit_cast(v16bf, bu1), (short)0, acc1, false, false);
    }
  }

  // ---- epilogue: C/D layout VGPR i -> row i+8*hi, lane%16 -> col ----
  const int row0 = blockIdx.x * 16;
  const int gc0 = ct0 * 16 + r, gc1 = ct1 * 16 + r;
  if (row0 + 16 <= nrows) {  // uniform fast path (always taken for N%16==0)
#pragma unroll
    for (int i = 0; i < 8; ++i) {
      int gr = row0 + i + 8 * hi;
      float v = acc0[i];
      if (bias) v += bias[gc0];
      if (relu) v = v > 0.f ? v : 0.f;
      Cout[(long)gr * OUTD + gc0] = v;
      if (two) {
        float w = acc1[i];
        if (bias) w += bias[gc1];
        if (relu) w = w > 0.f ? w : 0.f;
        Cout[(long)gr * OUTD + gc1] = w;
      }
    }
  } else {
#pragma unroll
    for (int i = 0; i < 8; ++i) {
      int gr = row0 + i + 8 * hi;
      if (gr < nrows) {
        float v = acc0[i];
        if (bias) v += bias[gc0];
        if (relu) v = v > 0.f ? v : 0.f;
        Cout[(long)gr * OUTD + gc0] = v;
        if (two) {
          float w = acc1[i];
          if (bias) w += bias[gc1];
          if (relu) w = w > 0.f ? w : 0.f;
          Cout[(long)gr * OUTD + gc1] = w;
        }
      }
    }
  }
}

// ---------------------------------------------------------------------------
// Generic fill
// ---------------------------------------------------------------------------
__global__ void fill_kernel(float* __restrict__ p, float v, long n) {
  long i = (long)blockIdx.x * blockDim.x + threadIdx.x;
  if (i < n) p[i] = v;
}

// ---------------------------------------------------------------------------
// mean(edge_attr) over E rows of 16 (accumulate then scale)
// ---------------------------------------------------------------------------
__global__ void mean_accum_kernel(const float* __restrict__ ea, float* __restrict__ acc, int E) {
  int t = blockIdx.x * blockDim.x + threadIdx.x;
  int stride = gridDim.x * blockDim.x;
  float s[16];
#pragma unroll
  for (int k = 0; k < 16; ++k) s[k] = 0.f;
  for (int e = t; e < E; e += stride) {
    const float* row = ea + (long)e * 16;
#pragma unroll
    for (int k = 0; k < 16; ++k) s[k] += row[k];
  }
#pragma unroll
  for (int k = 0; k < 16; ++k) atomicAdd(&acc[k], s[k]);
}

__global__ void scale16_kernel(float* __restrict__ p, float s) {
  int t = threadIdx.x;
  if (t < 16) p[t] *= s;
}

// ---------------------------------------------------------------------------
// V[k,h] = sum_c We[k, h*C+c] * ae[h,c]   (folds ea@We GEMM 16x down)
// ---------------------------------------------------------------------------
__global__ void vea_kernel(const float* __restrict__ We, const float* __restrict__ ae,
                           float* __restrict__ V, int H, int C) {
  int t = threadIdx.x;
  if (t < 16 * H) {
    int k = t / H, h = t % H;
    float s = 0.f;
    for (int c = 0; c < C; ++c) s += We[k * 128 + h * C + c] * ae[h * C + c];
    V[k * H + h] = s;
  }
}

// ---------------------------------------------------------------------------
// alpha_s / alpha_d per (node, head)
// ---------------------------------------------------------------------------
__global__ void alpha_sd_kernel(const float* __restrict__ xl,
                                const float* __restrict__ as_, const float* __restrict__ ad_,
                                float* __restrict__ als, float* __restrict__ ald,
                                int Nn, int H, int C) {
  long idx = (long)blockIdx.x * blockDim.x + threadIdx.x;
  if (idx >= (long)Nn * H) return;
  int n = (int)(idx / H), h = (int)(idx % H);
  const float* row = xl + (long)n * 128 + h * C;
  float ss = 0.f, sd = 0.f;
  for (int c = 0; c < C; ++c) {
    float v = row[c];
    ss += v * as_[h * C + c];
    sd += v * ad_[h * C + c];
  }
  als[idx] = ss;
  ald[idx] = sd;
}

// ---------------------------------------------------------------------------
// Pass 1: alpha = leakyrelu(as[src]+ad[dst]+ae), store; segment atomic-max
// ---------------------------------------------------------------------------
__global__ void edge_alpha_max_kernel(const int* __restrict__ ei, const float* __restrict__ eattr,
                                      const float* __restrict__ meanea, const float* __restrict__ V,
                                      const float* __restrict__ als, const float* __restrict__ ald,
                                      float* __restrict__ exbuf, float* __restrict__ segmax,
                                      int E, int Nn, int H) {
  long idx = (long)blockIdx.x * blockDim.x + threadIdx.x;
  long tot = (long)(E + Nn) * H;
  if (idx >= tot) return;
  int e = (int)(idx / H), h = (int)(idx % H);
  int s, d;
  const float* ea;
  if (e < E) { s = ei[e]; d = ei[E + e]; ea = eattr + (long)e * 16; }
  else       { s = d = e - E;            ea = meanea; }
  float aE = 0.f;
#pragma unroll
  for (int k = 0; k < 16; ++k) aE += ea[k] * V[k * H + h];
  float a = als[(long)s * H + h] + ald[(long)d * H + h] + aE;
  a = a > 0.f ? a : NEG_SLOPE * a;
  exbuf[idx] = a;
  atomicMaxF(&segmax[(long)d * H + h], a);
}

// ---------------------------------------------------------------------------
// Pass 2: ex = exp(alpha - max[dst]); segment atomic-sum
// ---------------------------------------------------------------------------
__global__ void edge_expsum_kernel(const int* __restrict__ ei,
                                   const float* __restrict__ segmax,
                                   float* __restrict__ exbuf, float* __restrict__ segsum,
                                   int E, int Nn, int H) {
  long idx = (long)blockIdx.x * blockDim.x + threadIdx.x;
  long tot = (long)(E + Nn) * H;
  if (idx >= tot) return;
  int e = (int)(idx / H), h = (int)(idx % H);
  int d = (e < E) ? ei[E + e] : e - E;
  float ex = __expf(exbuf[idx] - segmax[(long)d * H + h]);
  exbuf[idx] = ex;
  atomicAdd(&segsum[(long)d * H + h], ex);
}

// ---------------------------------------------------------------------------
// Pass 3: agg[dst, c] += xl[src, c] * ex/denom  (one thread per edge-channel)
// ---------------------------------------------------------------------------
__global__ void aggregate_kernel(const int* __restrict__ ei,
                                 const float* __restrict__ exbuf, const float* __restrict__ segsum,
                                 const float* __restrict__ xl, float* __restrict__ agg,
                                 int E, int Nn, int H, int C) {
  long idx = (long)blockIdx.x * blockDim.x + threadIdx.x;
  long tot = (long)(E + Nn) * 128;
  if (idx >= tot) return;
  int e = (int)(idx >> 7);
  int c = (int)(idx & 127);
  int h = c / C;
  int s, d;
  if (e < E) { s = ei[e]; d = ei[E + e]; }
  else       { s = d = e - E; }
  float w = exbuf[(long)e * H + h] / segsum[(long)d * H + h];
  atomicAdd(&agg[(long)d * 128 + c], xl[(long)s * 128 + c] * w);
}

// ---------------------------------------------------------------------------
// Host-side driver
// ---------------------------------------------------------------------------
namespace {

struct Scratch {
  unsigned short *xbf, *fb0, *fb1;          // bf16 A-fragment feature buffers
  unsigned short *W1b, *W2b, *W3b, *Wlb;    // bf16 B-fragment weights
  float *xl, *agg, *als, *ald, *segmax, *segsum, *exbuf, *Vea, *meanea;
};

inline unsigned blks(long n, int tpb) { return (unsigned)((n + tpb - 1) / tpb); }

inline void run_gat_layer(const unsigned short* in_bf, int Kin, int H, int C,
                          const unsigned short* Wb, const float* as_, const float* ad_,
                          const float* We, const float* ae_, const float* bb,
                          unsigned short* out_bf, const int* ei, const float* eattr,
                          int Nn, int E, const Scratch& S, hipStream_t stream) {
  const int TPB = 256;
  const long eTot = (long)E + Nn;

  // 1) xl = in @ W (bf16 WMMA, fp32 out, row-major for attention/aggregation)
  gemm_swz_bf16_wmma<<<(Nn + 15) / 16, 128, 0, stream>>>(
      in_bf, Wb, nullptr, S.xl, Nn, Kin, 128, 0);

  // 2) per-node attention logits
  alpha_sd_kernel<<<blks((long)Nn * H, TPB), TPB, 0, stream>>>(
      S.xl, as_, ad_, S.als, S.ald, Nn, H, C);

  // 3) edge-attention projection vectors V = We @ ae (16 x H)
  vea_kernel<<<1, 128, 0, stream>>>(We, ae_, S.Vea, H, C);

  // 4) init segment buffers + output accumulator
  fill_kernel<<<blks((long)Nn * H, TPB), TPB, 0, stream>>>(S.segmax, -1e30f, (long)Nn * H);
  fill_kernel<<<blks((long)Nn * H, TPB), TPB, 0, stream>>>(S.segsum, 0.f, (long)Nn * H);
  fill_kernel<<<blks((long)Nn * 128, TPB), TPB, 0, stream>>>(S.agg, 0.f, (long)Nn * 128);

  // 5-7) segment softmax + weighted scatter
  edge_alpha_max_kernel<<<blks(eTot * H, TPB), TPB, 0, stream>>>(
      ei, eattr, S.meanea, S.Vea, S.als, S.ald, S.exbuf, S.segmax, E, Nn, H);
  edge_expsum_kernel<<<blks(eTot * H, TPB), TPB, 0, stream>>>(
      ei, S.segmax, S.exbuf, S.segsum, E, Nn, H);
  aggregate_kernel<<<blks(eTot * 128, TPB), TPB, 0, stream>>>(
      ei, S.exbuf, S.segsum, S.xl, S.agg, E, Nn, H, C);

  // 8) bias + relu, emitted directly as bf16 A-fragment layout for next GEMM
  pack_feat_kernel<<<blks((long)Nn * 128, TPB), TPB, 0, stream>>>(
      S.agg, bb, out_bf, (long)Nn * 128, 128, 1);
}

}  // namespace

extern "C" void kernel_launch(void* const* d_in, const int* in_sizes, int n_in,
                              void* d_out, int out_size, void* d_ws, size_t ws_size,
                              hipStream_t stream) {
  const float* x     = (const float*)d_in[0];
  const int*   ei    = (const int*)d_in[1];
  const float* eattr = (const float*)d_in[2];
  const float* W1  = (const float*)d_in[3];
  const float* as1 = (const float*)d_in[4];
  const float* ad1 = (const float*)d_in[5];
  const float* We1 = (const float*)d_in[6];
  const float* ae1 = (const float*)d_in[7];
  const float* b1  = (const float*)d_in[8];
  const float* W2  = (const float*)d_in[9];
  const float* as2 = (const float*)d_in[10];
  const float* ad2 = (const float*)d_in[11];
  const float* We2 = (const float*)d_in[12];
  const float* ae2 = (const float*)d_in[13];
  const float* b2  = (const float*)d_in[14];
  const float* W3  = (const float*)d_in[15];
  const float* as3 = (const float*)d_in[16];
  const float* ad3 = (const float*)d_in[17];
  const float* We3 = (const float*)d_in[18];
  const float* ae3 = (const float*)d_in[19];
  const float* b3  = (const float*)d_in[20];
  const float* Wl  = (const float*)d_in[21];
  const float* bl  = (const float*)d_in[22];

  const int Nn = in_sizes[0] / 64;   // 50000
  const int E  = in_sizes[1] / 2;    // 1600000
  const long eTot = (long)E + Nn;

  // Carve workspace (256B-aligned chunks)
  char* base = (char*)d_ws;
  size_t off = 0;
  auto carveB = [&](size_t nbytes) -> void* {
    void* p = (void*)(base + off);
    off += (nbytes + 255) & ~(size_t)255;
    return p;
  };
  Scratch S;
  S.xbf    = (unsigned short*)carveB((size_t)Nn * 64 * 2);
  S.fb0    = (unsigned short*)carveB((size_t)Nn * 128 * 2);
  S.fb1    = (unsigned short*)carveB((size_t)Nn * 128 * 2);
  S.W1b    = (unsigned short*)carveB((size_t)64 * 128 * 2);
  S.W2b    = (unsigned short*)carveB((size_t)128 * 128 * 2);
  S.W3b    = (unsigned short*)carveB((size_t)128 * 128 * 2);
  S.Wlb    = (unsigned short*)carveB((size_t)128 * 64 * 2);
  S.xl     = (float*)carveB((size_t)Nn * 128 * 4);
  S.agg    = (float*)carveB((size_t)Nn * 128 * 4);
  S.als    = (float*)carveB((size_t)Nn * 8 * 4);
  S.ald    = (float*)carveB((size_t)Nn * 8 * 4);
  S.segmax = (float*)carveB((size_t)Nn * 8 * 4);
  S.segsum = (float*)carveB((size_t)Nn * 8 * 4);
  S.exbuf  = (float*)carveB((size_t)eTot * 8 * 4);
  S.Vea    = (float*)carveB(128 * 4);
  S.meanea = (float*)carveB(16 * 4);
  (void)ws_size;

  const int TPB = 256;

  // One-time packs (run every call; deterministic)
  pack_feat_kernel<<<blks((long)Nn * 64, TPB), TPB, 0, stream>>>(
      x, nullptr, S.xbf, (long)Nn * 64, 64, 0);
  pack_w_kernel<<<blks(64 * 128, TPB), TPB, 0, stream>>>(W1, S.W1b, 64, 128);
  pack_w_kernel<<<blks(128 * 128, TPB), TPB, 0, stream>>>(W2, S.W2b, 128, 128);
  pack_w_kernel<<<blks(128 * 128, TPB), TPB, 0, stream>>>(W3, S.W3b, 128, 128);
  pack_w_kernel<<<blks(128 * 64, TPB), TPB, 0, stream>>>(Wl, S.Wlb, 128, 64);

  // mean(edge_attr) once (self-loop edge features)
  fill_kernel<<<1, 32, 0, stream>>>(S.meanea, 0.f, 16);
  mean_accum_kernel<<<256, TPB, 0, stream>>>(eattr, S.meanea, E);
  scale16_kernel<<<1, 32, 0, stream>>>(S.meanea, 1.0f / (float)E);

  // conv1: in=64, H=8, C=16
  run_gat_layer(S.xbf, 64,  8, 16,  S.W1b, as1, ad1, We1, ae1, b1, S.fb0, ei, eattr, Nn, E, S, stream);
  // conv2: in=128, H=8, C=16
  run_gat_layer(S.fb0, 128, 8, 16,  S.W2b, as2, ad2, We2, ae2, b2, S.fb1, ei, eattr, Nn, E, S, stream);
  // conv3: in=128, H=1, C=128
  run_gat_layer(S.fb1, 128, 1, 128, S.W3b, as3, ad3, We3, ae3, b3, S.fb0, ei, eattr, Nn, E, S, stream);

  // final linear + relu -> d_out (N x 64, fp32)
  gemm_swz_bf16_wmma<<<(Nn + 15) / 16, 128, 0, stream>>>(
      S.fb0, S.Wlb, bl, (float*)d_out, Nn, 128, 64, 1);
}
